// QwenBlock_32083405701332
// MI455X (gfx1250) — compile-verified, matrix operand order checked by reading, not run on hardware
//
#include <hip/hip_runtime.h>
#include <stdint.h>

// ---------------------------------------------------------------------------
// Types for CDNA5 WMMA (wave32): v_wmma_f32_16x16x32_bf16
// ---------------------------------------------------------------------------
typedef __attribute__((ext_vector_type(16))) __bf16          bf16x16;
typedef __attribute__((ext_vector_type(8)))  float           v8f;
typedef __attribute__((ext_vector_type(8)))  unsigned short  ushort8;

union FragU { bf16x16 v; ushort8 h[2]; };

__device__ __forceinline__ unsigned short f2bf(float f) {
    union { float f; unsigned u; } c; c.f = f;
    unsigned u = c.u;
    return (unsigned short)((u + 0x7FFFu + ((u >> 16) & 1u)) >> 16); // RNE
}
__device__ __forceinline__ float bf2f(unsigned short h) {
    union { unsigned u; float f; } c; c.u = ((unsigned)h) << 16; return c.f;
}

__device__ __forceinline__ v8f wmma_bf16(const FragU& a, const FragU& b, v8f c) {
    return __builtin_amdgcn_wmma_f32_16x16x32_bf16(
        /*neg_a=*/false, a.v, /*neg_b=*/false, b.v,
        /*c_mod=*/(short)0, c, /*reuse_a=*/false, /*reuse_b=*/false);
}

// ---------------------------------------------------------------------------
// Tensor Data Mover staging: one 2D D# load of a 128-row x 32-col bf16 tile
// from a row-major [rows, K] tensor into LDS with a 40-ushort padded row
// stride (pad 4 DWORDs every 16 DWORDs).
// ---------------------------------------------------------------------------
#if defined(__gfx1250__) && __has_builtin(__builtin_amdgcn_tensor_load_to_lds)
#define USE_TDM 1
#else
#define USE_TDM 0
#endif

#if USE_TDM
typedef __attribute__((ext_vector_type(4))) unsigned int uint32x4;
typedef __attribute__((ext_vector_type(8))) int          int32x8;
typedef __attribute__((ext_vector_type(4))) int          int32x4;

__device__ __forceinline__ void tdm_load_tile_128x32(const unsigned short* gsrc,
                                                     unsigned int ldsOff,
                                                     unsigned int K) {
    const unsigned long long ga = (unsigned long long)(uintptr_t)gsrc;
    uint32x4 g0;
    g0.x = 1u;                                   // count=1, user mode
    g0.y = ldsOff;                               // lds_addr (bytes)
    g0.z = (unsigned int)ga;                     // global_addr[31:0]
    g0.w = (unsigned int)((ga >> 32) & 0x01FFFFFFu) | (2u << 30); // addr[56:32] | type=2
    int32x8 g1;
    g1[0] = (int)((1u << 16)        // data_size = 2 bytes
                | (1u << 20)        // pad_enable
                | (3u << 22)        // pad_interval: 16 DWORDs
                | (3u << 25));      // pad_amount : 4 DWORDs
    g1[1] = (int)((K & 0xFFFFu) << 16);           // tensor_dim0[15:0]
    g1[2] = (int)((K >> 16) | (0xFFFFu << 16));   // tensor_dim0[31:16] | tensor_dim1 lo
    g1[3] = (int)(32u << 16);                     // tensor_dim1 hi=0 | tile_dim0=32
    g1[4] = (int)128u;                            // tile_dim1=128, tile_dim2=0
    g1[5] = (int)K;                               // tensor_dim0_stride[31:0]
    g1[6] = 0;                                    // stride hi / dim1_stride lo
    g1[7] = 0;
    const int32x4 z4 = {0, 0, 0, 0};
#if __clang_major__ >= 23
    const int32x8 z8 = {0, 0, 0, 0, 0, 0, 0, 0};
    __builtin_amdgcn_tensor_load_to_lds(g0, g1, z4, z4, z8, 0);
#else
    __builtin_amdgcn_tensor_load_to_lds(g0, g1, z4, z4, 0);
#endif
}
#endif

// ---------------------------------------------------------------------------
// fp32 -> bf16 cast
// ---------------------------------------------------------------------------
__global__ void cast_f32_bf16(const float* __restrict__ src,
                              unsigned short* __restrict__ dst, int n) {
    int i = blockIdx.x * blockDim.x + threadIdx.x;
    int stride = gridDim.x * blockDim.x;
    for (; i < n; i += stride) dst[i] = f2bf(src[i]);
}

// ---------------------------------------------------------------------------
// RMSNorm: one block (256 thr) per row of D=2048, bf16 output
// ---------------------------------------------------------------------------
__global__ __launch_bounds__(256)
void rmsnorm_kernel(const float* __restrict__ x, const float* __restrict__ w,
                    unsigned short* __restrict__ out, int D) {
    const int row = blockIdx.x;
    const float* xr = x + (size_t)row * D;
    unsigned short* outr = out + (size_t)row * D;

    float ss = 0.f;
    for (int i = threadIdx.x; i < D; i += blockDim.x) { float v = xr[i]; ss += v * v; }
    #pragma unroll
    for (int m = 16; m >= 1; m >>= 1) ss += __shfl_xor(ss, m, 32);

    __shared__ float red[8];
    const int lane = threadIdx.x & 31, wid = threadIdx.x >> 5;
    if (lane == 0) red[wid] = ss;
    __syncthreads();
    if (wid == 0) {
        float v = (lane < 8) ? red[lane] : 0.f;
        #pragma unroll
        for (int m = 4; m >= 1; m >>= 1) v += __shfl_xor(v, m, 32);
        if (lane == 0) red[0] = v;
    }
    __syncthreads();
    const float norm = rsqrtf(red[0] / (float)D + 1e-6f);
    for (int i = threadIdx.x; i < D; i += blockDim.x)
        outr[i] = f2bf(xr[i] * norm * w[i]);
}

// ---------------------------------------------------------------------------
// WMMA GEMM: C[N,M] = A[N,K] (bf16) x W[M,K]^T (bf16) + bias (+residual)
// Block: 256 thr (8 waves). Block tile 128x128, K step 32.
// Wave (wid&3, wid>>2) computes a 32x64 patch = 2x4 WMMA tiles (8 WMMA/step).
// LDS tiles staged by the Tensor Data Mover (wave 0) when available.
// ---------------------------------------------------------------------------
#define GTN 128
#define GTM 128
#define GTK 32

__global__ __launch_bounds__(256)
void gemm_bf16_wmma(const unsigned short* __restrict__ A,
                    const unsigned short* __restrict__ W,
                    const float* __restrict__ bias,     // [M] or null
                    const float* __restrict__ resid,    // [N,M] or null
                    float* __restrict__ outF,           // [N,M] or null
                    unsigned short* __restrict__ outB,  // [N,M] or null
                    int N, int M, int K) {
    __shared__ unsigned short As[GTN][GTK + 8];
    __shared__ unsigned short Bs[GTM][GTK + 8];

    const int t = threadIdx.x;
    const int lane = t & 31, wid = t >> 5;
    const int nBase = blockIdx.y * GTN;
    const int mBase = blockIdx.x * GTM;
    const int r0 = (wid & 3) * 32;    // row band inside A tile (2 x 16)
    const int c0 = (wid >> 2) * 64;   // col band inside B tile (4 x 16)
    const int lrow = lane & 15;
    const int halfSel = (lane & 16) ? 8 : 0;
    const int hi8 = (lane >> 4) << 3;

    v8f acc[2][4] = {};

#if USE_TDM
    const unsigned int asOff = (unsigned int)(uintptr_t)(void*)&As[0][0];
    const unsigned int bsOff = (unsigned int)(uintptr_t)(void*)&Bs[0][0];
#else
    const int cA = t & 3, rA = t >> 2;  // 64 rows x 4 chunks of 8 per pass
#endif

    for (int k0 = 0; k0 < K; k0 += GTK) {
#if USE_TDM
        if (wid == 0) {
            tdm_load_tile_128x32(A + (size_t)nBase * K + k0, asOff, (unsigned)K);
            tdm_load_tile_128x32(W + (size_t)mBase * K + k0, bsOff, (unsigned)K);
            __builtin_amdgcn_s_wait_tensorcnt(0);
        }
        __syncthreads();
#else
        *(ushort8*)&As[rA][cA * 8] =
            *(const ushort8*)(A + (size_t)(nBase + rA) * K + k0 + cA * 8);
        *(ushort8*)&As[64 + rA][cA * 8] =
            *(const ushort8*)(A + (size_t)(nBase + 64 + rA) * K + k0 + cA * 8);
        *(ushort8*)&Bs[rA][cA * 8] =
            *(const ushort8*)(W + (size_t)(mBase + rA) * K + k0 + cA * 8);
        *(ushort8*)&Bs[64 + rA][cA * 8] =
            *(const ushort8*)(W + (size_t)(mBase + 64 + rA) * K + k0 + cA * 8);
        __syncthreads();
#endif

        FragU a[2], b[4];
        #pragma unroll
        for (int i = 0; i < 2; i++) {
            const unsigned short* p = &As[r0 + i * 16 + lrow][halfSel];
            a[i].h[0] = *(const ushort8*)p;
            a[i].h[1] = *(const ushort8*)(p + 16);
        }
        #pragma unroll
        for (int j = 0; j < 4; j++) {
            const unsigned short* p = &Bs[c0 + j * 16 + lrow][halfSel];
            b[j].h[0] = *(const ushort8*)p;
            b[j].h[1] = *(const ushort8*)(p + 16);
        }
        #pragma unroll
        for (int i = 0; i < 2; i++)
            #pragma unroll
            for (int j = 0; j < 4; j++)
                acc[i][j] = wmma_bf16(a[i], b[j], acc[i][j]);
        __syncthreads();
    }

    // epilogue: C/D tile layout -> row m = g + 8*(lane/16), col n = lane%16
    #pragma unroll
    for (int i = 0; i < 2; i++)
        #pragma unroll
        for (int j = 0; j < 4; j++)
            #pragma unroll
            for (int g = 0; g < 8; g++) {
                const int row = nBase + r0 + i * 16 + g + hi8;
                const int col = mBase + c0 + j * 16 + lrow;
                float v = acc[i][j][g];
                if (bias)  v += bias[col];
                if (resid) v += resid[(size_t)row * M + col];
                if (outF)  outF[(size_t)row * M + col] = v;
                else       outB[(size_t)row * M + col] = f2bf(v);
            }
}

// ---------------------------------------------------------------------------
// Flash attention (causal, GQA h -> h%4), bf16 QKV, fp32 online softmax.
// Block: 128 thr (4 waves). Each block: one (b, h, 64-row q tile).
// Each wave owns 16 q rows; K/V streamed in 32-row chunks through LDS.
// ---------------------------------------------------------------------------
#define FA_KC 32

__global__ __launch_bounds__(128)
void flash_attn_kernel(const unsigned short* __restrict__ Q,  // [B*L, 2048]
                       const unsigned short* __restrict__ Kb, // [B*L, 512]
                       const unsigned short* __restrict__ Vb, // [B*L, 512]
                       unsigned short* __restrict__ O,        // [B*L, 2048]
                       int L) {
    __shared__ unsigned short Ks[FA_KC][128 + 8];     // [k][hd]
    __shared__ unsigned short Vt[128][FA_KC + 8];     // [hd][k] (transposed)
    __shared__ unsigned short Ps[4][16][32 + 8];      // per-wave P staging

    const int tid = threadIdx.x;
    const int lane = tid & 31, wq = tid >> 5;
    const int qi = blockIdx.x, h = blockIdx.y, b = blockIdx.z;
    const int kvh = h & 3;
    const int q0 = qi * 64;
    const int lrow = lane & 15;
    const int halfSel = (lane & 16) ? 8 : 0;
    const int hi8 = (lane >> 4) << 3;
    const float scale = 0.08838834764831845f;  // 128^-0.5

    // Q fragments for this wave's 16 rows (A-layout, direct from global)
    FragU qf[4];
    {
        const unsigned short* qp =
            Q + ((size_t)(b * L + q0 + wq * 16 + lrow) * 2048) + h * 128;
        #pragma unroll
        for (int hc = 0; hc < 4; hc++) {
            qf[hc].h[0] = *(const ushort8*)(qp + hc * 32 + halfSel);
            qf[hc].h[1] = *(const ushort8*)(qp + hc * 32 + halfSel + 16);
        }
    }

    float mrow[8], lrowsum[8];
    v8f o_acc[8] = {};
    #pragma unroll
    for (int g = 0; g < 8; g++) { mrow[g] = -1e30f; lrowsum[g] = 0.f; }

    const int nchunks = qi * 2 + 2;  // all k chunks with k <= q0+63
    for (int ch = 0; ch < nchunks; ch++) {
        const int kc = ch * FA_KC;
        // stage K chunk [32x128] and V chunk transposed [128x32]
        #pragma unroll
        for (int i = 0; i < 4; i++) {
            const int lin = tid + i * 128;
            const int r = lin >> 4, c = lin & 15;
            const size_t gro = (size_t)(b * L + kc + r) * 512 + kvh * 128 + c * 8;
            ushort8 kv = *(const ushort8*)(Kb + gro);
            *(ushort8*)&Ks[r][c * 8] = kv;
            ushort8 vv = *(const ushort8*)(Vb + gro);
            #pragma unroll
            for (int e = 0; e < 8; e++) Vt[c * 8 + e][r] = vv[e];
        }
        __syncthreads();

        // S = Q * K^T  (two 16x16 score tiles per wave)
        v8f s0 = {}, s1 = {};
        #pragma unroll
        for (int hc = 0; hc < 4; hc++) {
            FragU bk0, bk1;
            const unsigned short* p0 = &Ks[lrow][hc * 32 + halfSel];
            bk0.h[0] = *(const ushort8*)p0;
            bk0.h[1] = *(const ushort8*)(p0 + 16);
            const unsigned short* p1 = &Ks[16 + lrow][hc * 32 + halfSel];
            bk1.h[0] = *(const ushort8*)p1;
            bk1.h[1] = *(const ushort8*)(p1 + 16);
            s0 = wmma_bf16(qf[hc], bk0, s0);
            s1 = wmma_bf16(qf[hc], bk1, s1);
        }

        // causal mask + online softmax update (per row g+hi8; 16-lane rows)
        float alpha[8];
        #pragma unroll
        for (int g = 0; g < 8; g++) {
            const int qg = q0 + wq * 16 + g + hi8;
            float v0 = s0[g] * scale, v1 = s1[g] * scale;
            if (kc + lrow > qg)      v0 = -1e30f;
            if (kc + 16 + lrow > qg) v1 = -1e30f;
            float rm = fmaxf(v0, v1);
            #pragma unroll
            for (int msk = 8; msk >= 1; msk >>= 1)
                rm = fmaxf(rm, __shfl_xor(rm, msk, 32));
            const float mn = fmaxf(mrow[g], rm);
            const float a = __expf(mrow[g] - mn);
            const float p0 = __expf(v0 - mn), p1 = __expf(v1 - mn);
            float rs = p0 + p1;
            #pragma unroll
            for (int msk = 8; msk >= 1; msk >>= 1)
                rs += __shfl_xor(rs, msk, 32);
            mrow[g] = mn;
            lrowsum[g] = lrowsum[g] * a + rs;
            alpha[g] = a;
            s0[g] = p0; s1[g] = p1;
        }
        #pragma unroll
        for (int tt = 0; tt < 8; tt++)
            #pragma unroll
            for (int g = 0; g < 8; g++) o_acc[tt][g] *= alpha[g];

        // P (C-layout) -> LDS -> A-layout bf16 fragment
        #pragma unroll
        for (int g = 0; g < 8; g++) {
            Ps[wq][g + hi8][lrow]      = f2bf(s0[g]);
            Ps[wq][g + hi8][16 + lrow] = f2bf(s1[g]);
        }
        __syncthreads();
        FragU pf;
        {
            const unsigned short* pp = &Ps[wq][lrow][halfSel];
            pf.h[0] = *(const ushort8*)pp;
            pf.h[1] = *(const ushort8*)(pp + 16);
        }
        // O += P * V  (8 hd tiles of 16)
        #pragma unroll
        for (int tt = 0; tt < 8; tt++) {
            FragU bv;
            const unsigned short* vp = &Vt[tt * 16 + lrow][halfSel];
            bv.h[0] = *(const ushort8*)vp;
            bv.h[1] = *(const ushort8*)(vp + 16);
            o_acc[tt] = wmma_bf16(pf, bv, o_acc[tt]);
        }
        __syncthreads();
    }

    // normalize and write out [token, h*128 + hd] as bf16
    #pragma unroll
    for (int tt = 0; tt < 8; tt++)
        #pragma unroll
        for (int g = 0; g < 8; g++) {
            const int row = q0 + wq * 16 + g + hi8;
            O[(size_t)(b * L + row) * 2048 + h * 128 + tt * 16 + lrow] =
                f2bf(o_acc[tt][g] / lrowsum[g]);
        }
}

// ---------------------------------------------------------------------------
// SwiGLU: g <- bf16( silu(g) * u )
// ---------------------------------------------------------------------------
__global__ void swiglu_kernel(unsigned short* __restrict__ g,
                              const unsigned short* __restrict__ u, int n) {
    int i = blockIdx.x * blockDim.x + threadIdx.x;
    int stride = gridDim.x * blockDim.x;
    for (; i < n; i += stride) {
        const float gv = bf2f(g[i]), uv = bf2f(u[i]);
        const float s = gv / (1.f + __expf(-gv));
        g[i] = f2bf(s * uv);
    }
}

// ---------------------------------------------------------------------------
// Host orchestration
// ---------------------------------------------------------------------------
extern "C" void kernel_launch(void* const* d_in, const int* in_sizes, int n_in,
                              void* d_out, int out_size, void* d_ws, size_t ws_size,
                              hipStream_t stream) {
    const float* x      = (const float*)d_in[0];
    const float* ln1_w  = (const float*)d_in[1];
    const float* q_w    = (const float*)d_in[2];
    const float* q_b    = (const float*)d_in[3];
    const float* k_w    = (const float*)d_in[4];
    const float* k_b    = (const float*)d_in[5];
    const float* v_w    = (const float*)d_in[6];
    const float* v_b    = (const float*)d_in[7];
    const float* o_w    = (const float*)d_in[8];
    const float* ln2_w  = (const float*)d_in[9];
    const float* gate_w = (const float*)d_in[10];
    const float* up_w   = (const float*)d_in[11];
    const float* down_w = (const float*)d_in[12];
    float* out = (float*)d_out;

    const int B = 2, L = 2048, D = 2048, F = 5632, KD = 512;
    const int N = B * L;

    char* ws = (char*)d_ws;
    size_t off = 0;
    auto alloc = [&](size_t bytes) -> void* {
        void* p = ws + off;
        off += (bytes + 255) & ~(size_t)255;
        return p;
    };
    unsigned short* wq_bf = (unsigned short*)alloc((size_t)D * D * 2);
    unsigned short* wk_bf = (unsigned short*)alloc((size_t)KD * D * 2);
    unsigned short* wv_bf = (unsigned short*)alloc((size_t)KD * D * 2);
    unsigned short* wo_bf = (unsigned short*)alloc((size_t)D * D * 2);
    unsigned short* wg_bf = (unsigned short*)alloc((size_t)F * D * 2);
    unsigned short* wu_bf = (unsigned short*)alloc((size_t)F * D * 2);
    unsigned short* wd_bf = (unsigned short*)alloc((size_t)D * F * 2);
    unsigned short* h1    = (unsigned short*)alloc((size_t)N * D * 2);
    unsigned short* qb    = (unsigned short*)alloc((size_t)N * D * 2);
    unsigned short* kb    = (unsigned short*)alloc((size_t)N * KD * 2);
    unsigned short* vb    = (unsigned short*)alloc((size_t)N * KD * 2);
    unsigned short* attn  = (unsigned short*)alloc((size_t)N * D * 2);
    float*          x2    = (float*)alloc((size_t)N * D * 4);
    unsigned short* h2    = (unsigned short*)alloc((size_t)N * D * 2);
    unsigned short* gbuf  = (unsigned short*)alloc((size_t)N * F * 2);
    unsigned short* ubuf  = (unsigned short*)alloc((size_t)N * F * 2);
    (void)ws_size; (void)n_in; (void)in_sizes; (void)out_size;

    // 1) cast weights to bf16
    cast_f32_bf16<<<2048, 256, 0, stream>>>(q_w,    wq_bf, D * D);
    cast_f32_bf16<<<2048, 256, 0, stream>>>(k_w,    wk_bf, KD * D);
    cast_f32_bf16<<<2048, 256, 0, stream>>>(v_w,    wv_bf, KD * D);
    cast_f32_bf16<<<2048, 256, 0, stream>>>(o_w,    wo_bf, D * D);
    cast_f32_bf16<<<4096, 256, 0, stream>>>(gate_w, wg_bf, F * D);
    cast_f32_bf16<<<4096, 256, 0, stream>>>(up_w,   wu_bf, F * D);
    cast_f32_bf16<<<4096, 256, 0, stream>>>(down_w, wd_bf, D * F);

    // 2) attention sub-block
    rmsnorm_kernel<<<N, 256, 0, stream>>>(x, ln1_w, h1, D);
    gemm_bf16_wmma<<<dim3(D / GTM, N / GTN), 256, 0, stream>>>(
        h1, wq_bf, q_b, nullptr, nullptr, qb, N, D, D);
    gemm_bf16_wmma<<<dim3(KD / GTM, N / GTN), 256, 0, stream>>>(
        h1, wk_bf, k_b, nullptr, nullptr, kb, N, KD, D);
    gemm_bf16_wmma<<<dim3(KD / GTM, N / GTN), 256, 0, stream>>>(
        h1, wv_bf, v_b, nullptr, nullptr, vb, N, KD, D);
    flash_attn_kernel<<<dim3(L / 64, 16, B), 128, 0, stream>>>(qb, kb, vb, attn, L);
    gemm_bf16_wmma<<<dim3(D / GTM, N / GTN), 256, 0, stream>>>(
        attn, wo_bf, nullptr, x, x2, nullptr, N, D, D);

    // 3) MLP sub-block
    rmsnorm_kernel<<<N, 256, 0, stream>>>(x2, ln2_w, h2, D);
    gemm_bf16_wmma<<<dim3(F / GTM, N / GTN), 256, 0, stream>>>(
        h2, wg_bf, nullptr, nullptr, nullptr, gbuf, N, F, D);
    gemm_bf16_wmma<<<dim3(F / GTM, N / GTN), 256, 0, stream>>>(
        h2, wu_bf, nullptr, nullptr, nullptr, ubuf, N, F, D);
    swiglu_kernel<<<4096, 256, 0, stream>>>(gbuf, ubuf, N * F);
    gemm_bf16_wmma<<<dim3(D / GTM, N / GTN), 256, 0, stream>>>(
        gbuf, wd_bf, nullptr, x2, out, nullptr, N, D, F);
}
